// CausalSelfAttention_88897233092910
// MI455X (gfx1250) — compile-verified, matrix operand order checked by reading, not run on hardware
//
#include <hip/hip_runtime.h>
#include <hip/hip_bf16.h>
#include <stdint.h>

typedef __attribute__((ext_vector_type(16))) __bf16 v16bf;
typedef __attribute__((ext_vector_type(8)))  __bf16 v8bf;
typedef __attribute__((ext_vector_type(8)))  float  v8f;

#define WMMA_BF16(a,b,c) __builtin_amdgcn_wmma_f32_16x16x32_bf16(false,(a),false,(b),(short)0,(c),false,false)

// native bf16 converts (gfx1250 has hardware cvt; RNE)
static __device__ __forceinline__ __bf16 f2bf(float f){ return (__bf16)f; }
static __device__ __forceinline__ float bf2f(__bf16 b){ return (float)b; }

// Per-lane async copy: 16B global -> LDS, tracked by ASYNCcnt (CDNA5).
static __device__ __forceinline__ void async_b128(uint32_t lds_off, uint64_t sbase, uint32_t goff){
  asm volatile("global_load_async_to_lds_b128 %0, %1, %2"
               :: "v"(lds_off), "v"(goff), "s"(sbase) : "memory");
}
static __device__ __forceinline__ void wait_async0(){
  asm volatile("s_wait_asynccnt 0" ::: "memory");
}
static __device__ __forceinline__ uint32_t lds_off(const void* p){
  return (uint32_t)(uintptr_t)p;   // low 32 bits of generic ptr = LDS offset
}

// Load a 16-half WMMA fragment: elements e=0..7 at p[0..7], e=8..15 at p[16..23]
// (ISA 16-bit A/B layout when p = row_base + (lane>>4)*8).
static __device__ __forceinline__ v16bf frag_ld(const __bf16* p){
  v8bf lo = *(const v8bf*)p;
  v8bf hi = *(const v8bf*)(p+16);
  v16bf r;
#pragma unroll
  for (int i=0;i<8;i++){ r[i]=lo[i]; r[i+8]=hi[i]; }
  return r;
}

// ---------------------------------------------------------------- converts
__global__ void cvt_f32_bf16(const float* __restrict__ in, __bf16* __restrict__ out, int n){
  int i = (blockIdx.x*blockDim.x + threadIdx.x)*4;
  if (i < n){
    float4 v = *(const float4*)(in+i);
    out[i+0]=f2bf(v.x); out[i+1]=f2bf(v.y); out[i+2]=f2bf(v.z); out[i+3]=f2bf(v.w);
  }
}

// in[K][N] fp32 -> out[N][K] bf16, 32x32 LDS tile transpose
__global__ void cvt_transpose_bf16(const float* __restrict__ in, __bf16* __restrict__ out,
                                   int K, int N){
  __shared__ __bf16 t[32][33];
  const int k0 = blockIdx.y*32, n0 = blockIdx.x*32;
  const int tx = threadIdx.x & 31, ty = threadIdx.x >> 5;   // ty 0..7
#pragma unroll
  for (int i=0;i<4;i++){
    const int k = ty + i*8;
    t[k][tx] = f2bf(in[(size_t)(k0+k)*N + n0 + tx]);
  }
  __syncthreads();
#pragma unroll
  for (int i=0;i<4;i++){
    const int n = ty + i*8;
    out[(size_t)(n0+n)*K + k0 + tx] = t[tx][n];
  }
}

// ---------------------------------------------------------------- GEMM
// C[M,N] = A[M,K] * BT[N,K]^T (+bias); A,BT bf16. Block tile 128x128, K-step 32,
// 8 waves, async double-buffered LDS staging.
// MODE 0: scatter bf16 into q/k/v (q,k: [B,H,S,D]; v: [B,H,D,S]).
// MODE 1: fp32 C + bias.
template<int MODE>
__global__ void __launch_bounds__(256)
gemm_wmma(const __bf16* __restrict__ A, const __bf16* __restrict__ BT,
          const float* __restrict__ bias,
          __bf16* __restrict__ qout, __bf16* __restrict__ kout, __bf16* __restrict__ vout,
          float* __restrict__ Cout, int M, int N, int K)
{
  __shared__ __attribute__((aligned(16))) __bf16 As[2][128*40]; // row-major [m][k], pad 40
  __shared__ __attribute__((aligned(16))) __bf16 Bs[2][128*40]; // [n][k], pad 40

  const int tid  = threadIdx.x;
  const int lane = tid & 31;
  const int wv   = tid >> 5;
  const int wr   = wv & 3;            // wave M quadrant (32 rows)
  const int wc   = wv >> 2;           // wave N half (64 cols)
  const int lm   = lane & 15, lh = lane >> 4;
  const int blockM = blockIdx.y * 128;
  const int blockN = blockIdx.x * 128;

  const int srow = tid >> 1;            // 0..127 staging row
  const int sc16 = (tid & 1) * 16;      // halves chunk 0/16

  const uint64_t aB = (uint64_t)(uintptr_t)A;
  const uint64_t bB = (uint64_t)(uintptr_t)BT;

  auto stage = [&](int buf, int k0){
    uint32_t al = lds_off(&As[buf][srow*40 + sc16]);
    uint32_t ag = (uint32_t)(((size_t)(blockM+srow)*K + k0 + sc16)*2);
    async_b128(al,      aB, ag);
    async_b128(al + 16, aB, ag + 16);
    uint32_t bl = lds_off(&Bs[buf][srow*40 + sc16]);
    uint32_t bg = (uint32_t)(((size_t)(blockN+srow)*K + k0 + sc16)*2);
    async_b128(bl,      bB, bg);
    async_b128(bl + 16, bB, bg + 16);
  };

  v8f acc[2][4];
#pragma unroll
  for (int i=0;i<2;i++)
#pragma unroll
    for (int j=0;j<4;j++)
#pragma unroll
      for (int e=0;e<8;e++) acc[i][j][e] = 0.f;

  const int nK = K >> 5;
  stage(0, 0);
  for (int i=0; i<nK; i++){
    const int cur = i & 1, nxt = cur ^ 1;
    wait_async0();
    __syncthreads();
    if (i+1 < nK) stage(nxt, (i+1)*32);

    const __bf16* abase = &As[cur][(wr*32 + lm)*40 + lh*8];
    const __bf16* bbase = &Bs[cur][(wc*64 + lm)*40 + lh*8];
    v16bf a0 = frag_ld(abase);
    v16bf a1 = frag_ld(abase + 16*40);
#pragma unroll
    for (int ni=0; ni<4; ni++){
      v16bf bf = frag_ld(bbase + ni*16*40);
      acc[0][ni] = WMMA_BF16(a0, bf, acc[0][ni]);
      acc[1][ni] = WMMA_BF16(a1, bf, acc[1][ni]);
    }
  }

  // epilogue (C layout: lane = N + 16*(M>=8); vgpr r -> M = r + 8*(lane>>4))
  const int rbase = blockM + wr*32 + 8*lh;
  const int cbase = blockN + wc*64 + lm;
#pragma unroll
  for (int mi=0; mi<2; mi++){
#pragma unroll
    for (int ni=0; ni<4; ni++){
      const int col = cbase + ni*16;
      const float bv = bias[col];
#pragma unroll
      for (int r=0;r<8;r++){
        const int row = rbase + mi*16 + r;
        const float val = acc[mi][ni][r] + bv;
        if (MODE==1){
          Cout[(size_t)row*N + col] = val;
        } else {
          const int which = col >> 10;
          const int c = col & 1023;
          const int h = c >> 6, d = c & 63;
          const int b_ = row >> 11, s = row & 2047;
          if (which==2){ // V transposed: [B,H,D,S]
            vout[((size_t)(b_*16 + h)*64 + d)*2048 + s] = f2bf(val);
          } else {
            __bf16* dst = (which==0) ? qout : kout;
            dst[(((size_t)(b_*16 + h))*2048 + s)*64 + d] = f2bf(val);
          }
        }
      }
    }
  }
}

// ---------------------------------------------------------------- attention
// Grid: (S/64, B*H). Block 128 = 4 waves; wave handles 16 query rows.
// Q,K: [B,H,S,D] bf16; V: [B,H,D,S] bf16 (pre-transposed). Row sums via ones-column WMMA.
__global__ void __launch_bounds__(128)
attn_kernel(const __bf16* __restrict__ Q, const __bf16* __restrict__ Km,
            const __bf16* __restrict__ V, __bf16* __restrict__ O)
{
  __shared__ __attribute__((aligned(16))) __bf16 Ks[2][32*72];  // [key][d], pad 72
  __shared__ __attribute__((aligned(16))) __bf16 Vs[2][64*40];  // [d][key], pad 40
  __shared__ __attribute__((aligned(16))) __bf16 Vones[16*40];  // cols 64..79: col64=1, rest 0
  __shared__ __attribute__((aligned(16))) __bf16 Ps[4][16*40];  // per-wave P staging

  const int tid  = threadIdx.x;
  const int lane = tid & 31, wv = tid >> 5;
  const int lm   = lane & 15, lh = lane >> 4;
  const int q0   = blockIdx.x * 64;
  const int bh   = blockIdx.y;                  // b*16 + h
  const size_t base = (size_t)bh * 2048 * 64;
  const __bf16* Qp = Q + base;
  const uint64_t kB = (uint64_t)(uintptr_t)(Km + base);
  const uint64_t vB = (uint64_t)(uintptr_t)(V  + base);
  const int qRow = q0 + wv*16;

  // init ones/zero block for row-sum column (d=64 -> ones)
  {
#pragma unroll
    for (int j=0;j<4;j++){
      const int idx = tid*4 + j;            // 0..511
      const int dr = idx >> 5, kk = idx & 31;
      Vones[dr*40 + kk] = f2bf(dr==0 ? 1.0f : 0.0f);
    }
  }

  // Q fragments (A-layout over d), softmax scale 1/8 folded in (exact in bf16)
  v16bf aq[2];
#pragma unroll
  for (int c=0;c<2;c++){
    v16bf f = frag_ld(Qp + (size_t)(qRow+lm)*64 + c*32 + lh*8);
#pragma unroll
    for (int e=0;e<16;e++) f[e] = f2bf(bf2f(f[e]) * 0.125f);
    aq[c] = f;
  }

  v8f acc[5];                         // 4 output d-chunks + 1 row-sum chunk
#pragma unroll
  for (int c=0;c<5;c++)
#pragma unroll
    for (int r=0;r<8;r++) acc[c][r]=0.f;
  float mrow[8];
#pragma unroll
  for (int r=0;r<8;r++) mrow[r] = -3.0e38f;

  const int krow = tid >> 2, kc16 = (tid & 3) * 16;  // K staging: 32 rows x 128B
  const int drow = tid >> 1, vc16 = (tid & 1) * 16;  // V staging: 64 rows x 64B
  auto stage = [&](int buf, int kS){
    uint32_t kl = lds_off(&Ks[buf][krow*72 + kc16]);
    uint32_t kg = (uint32_t)(((size_t)(kS + krow)*64 + kc16)*2);
    async_b128(kl,      kB, kg);
    async_b128(kl + 16, kB, kg + 16);
    uint32_t vl = lds_off(&Vs[buf][drow*40 + vc16]);
    uint32_t vg = (uint32_t)(((size_t)drow*2048 + kS + vc16)*2);
    async_b128(vl,      vB, vg);
    async_b128(vl + 16, vB, vg + 16);
  };

  const int nIter = blockIdx.x*2 + 2;  // causal: keys up to q0+63
  stage(0, 0);
  for (int kb=0; kb<nIter; kb++){
    const int cur = kb & 1, nxt = cur ^ 1;
    const int kS = kb*32;
    wait_async0();
    __syncthreads();
    if (kb+1 < nIter) stage(nxt, (kb+1)*32);

    // Wave-uniform: skip tiles entirely above the diagonal for this wave's rows.
    if (kS <= qRow + 15){
      // scores S = Q K^T (two 16x16 column tiles, contracted over d=64)
      v8f s0, s1;
#pragma unroll
      for (int r=0;r<8;r++){ s0[r]=0.f; s1[r]=0.f; }
#pragma unroll
      for (int c=0;c<2;c++){
        v16bf bk0 = frag_ld(&Ks[cur][lm*72      + c*32 + lh*8]);
        v16bf bk1 = frag_ld(&Ks[cur][(16+lm)*72 + c*32 + lh*8]);
        s0 = WMMA_BF16(aq[c], bk0, s0);
        s1 = WMMA_BF16(aq[c], bk1, s1);
      }

      // causal mask only on diagonal-straddling tiles (wave-uniform test)
      const int qi0 = qRow + 8*lh;
      if (kS + 31 > qRow){
        const int k0i = kS + lm, k1i = kS + 16 + lm;
#pragma unroll
        for (int r=0;r<8;r++){
          const int qi = qi0 + r;
          if (k0i > qi) s0[r] = -3.0e38f;
          if (k1i > qi) s1[r] = -3.0e38f;
        }
      }

      // online softmax (rows r -> q = qi0 + r; cols = lm, 16+lm)
      float t[8];
#pragma unroll
      for (int r=0;r<8;r++) t[r] = fmaxf(s0[r], s1[r]);
#pragma unroll
      for (int msk=1; msk<16; msk<<=1)
#pragma unroll
        for (int r=0;r<8;r++) t[r] = fmaxf(t[r], __shfl_xor(t[r], msk, 32));

      float p0[8], p1[8];
#pragma unroll
      for (int r=0;r<8;r++){
        const float mn = fmaxf(mrow[r], t[r]);
        const float alpha = __expf(mrow[r] - mn);
        mrow[r] = mn;
        p0[r] = __expf(s0[r] - mn);
        p1[r] = __expf(s1[r] - mn);
#pragma unroll
        for (int c=0;c<5;c++) acc[c][r] *= alpha;   // rescale output + row-sum column
      }

      // restage P (score C-layout -> A-fragment layout) through LDS
      __bf16* Pw = &Ps[wv][0];
#pragma unroll
      for (int r=0;r<8;r++){
        Pw[(8*lh + r)*40 + lm     ] = f2bf(p0[r]);
        Pw[(8*lh + r)*40 + 16 + lm] = f2bf(p1[r]);
      }
      asm volatile("s_wait_dscnt 0" ::: "memory"); // wave-internal DS write->read ordering
      v16bf ap = frag_ld(&Pw[lm*40 + lh*8]);
#pragma unroll
      for (int c=0;c<4;c++){
        v16bf bv = frag_ld(&Vs[cur][(c*16+lm)*40 + lh*8]);
        acc[c] = WMMA_BF16(ap, bv, acc[c]);
      }
      { // ones-column: acc[4] column 0 accumulates row sums (rescaled with alpha above)
        v16bf b1 = frag_ld(&Vones[lm*40 + lh*8]);
        acc[4] = WMMA_BF16(ap, b1, acc[4]);
      }
    }
  }

  // epilogue: O[B,S,C] bf16, normalize by row sum (broadcast from lane lm==0 of each half)
  const int h = bh & 15, b_ = bh >> 4;
  float inv[8];
#pragma unroll
  for (int r=0;r<8;r++){
    const float l = __shfl(acc[4][r], lh*16, 32);
    inv[r] = 1.0f / l;
  }
#pragma unroll
  for (int c=0;c<4;c++)
#pragma unroll
    for (int r=0;r<8;r++){
      const int s = qRow + 8*lh + r;
      O[((size_t)(b_*2048 + s))*1024 + h*64 + c*16 + lm] = f2bf(acc[c][r]*inv[r]);
    }
}

// ---------------------------------------------------------------- launch
extern "C" void kernel_launch(void* const* d_in, const int* in_sizes, int n_in,
                              void* d_out, int out_size, void* d_ws, size_t ws_size,
                              hipStream_t stream)
{
  (void)in_sizes; (void)n_in; (void)out_size; (void)ws_size;
  const float* x      = (const float*)d_in[0];
  const float* w_qkv  = (const float*)d_in[1];
  const float* b_qkv  = (const float*)d_in[2];
  const float* w_proj = (const float*)d_in[3];
  const float* b_proj = (const float*)d_in[4];
  float* out = (float*)d_out;

  char* ws = (char*)d_ws;
  size_t off = 0;
  auto alloc = [&](size_t bytes)->void*{
    void* p = ws + off; off += (bytes + 255) & ~(size_t)255; return p;
  };
  __bf16* x_bf     = (__bf16*)alloc(8192ull*1024*2);   // 16 MB  [B*S, C]
  __bf16* wqkvT_bf = (__bf16*)alloc(3072ull*1024*2);   // 6 MB   [3C, C]
  __bf16* wprojT_bf= (__bf16*)alloc(1024ull*1024*2);   // 2 MB   [C, C]
  __bf16* q_bf     = (__bf16*)alloc(8192ull*1024*2);   // 16 MB  [B,H,S,D]
  __bf16* k_bf     = (__bf16*)alloc(8192ull*1024*2);   // 16 MB  [B,H,S,D]
  __bf16* v_bf     = (__bf16*)alloc(8192ull*1024*2);   // 16 MB  [B,H,D,S]
  __bf16* attn_bf  = (__bf16*)alloc(8192ull*1024*2);   // 16 MB  [B,S,C]

  { // conversions: x plain; weights transposed to [N][K]
    const int nx = 8192*1024;
    cvt_f32_bf16<<<nx/4/256, 256, 0, stream>>>(x, x_bf, nx);
    cvt_transpose_bf16<<<dim3(3072/32, 1024/32), 256, 0, stream>>>(w_qkv, wqkvT_bf, 1024, 3072);
    cvt_transpose_bf16<<<dim3(1024/32, 1024/32), 256, 0, stream>>>(w_proj, wprojT_bf, 1024, 1024);
  }
  // QKV projection: [8192,1024] x [1024,3072] -> scatter q/k/v bf16
  gemm_wmma<0><<<dim3(3072/128, 8192/128), 256, 0, stream>>>(
      x_bf, wqkvT_bf, b_qkv, q_bf, k_bf, v_bf, nullptr, 8192, 3072, 1024);
  // causal flash attention
  attn_kernel<<<dim3(2048/64, 64), 128, 0, stream>>>(q_bf, k_bf, v_bf, attn_bf);
  // output projection: [8192,1024] x [1024,1024] -> fp32 out
  gemm_wmma<1><<<dim3(1024/128, 8192/128), 256, 0, stream>>>(
      attn_bf, wprojT_bf, b_proj, nullptr, nullptr, nullptr, out, 8192, 1024, 1024);
}
